// MambaBlock_81956565943051
// MI455X (gfx1250) — matrix operand order repaired and checked by reading, not compiled, not run on hardware
//
#include <hip/hip_runtime.h>

// ---------------------------------------------------------------------------
// Mamba block for MI455X (gfx1250, wave32, WMMA), bf16 tensor path.
//   C0 cvt: x, W_in, W_x, W_dt, W_out  f32 -> bf16   (one streaming pass)
//   K1 gemm<1>: xr = x @ W_in      -> xs(f32) | silu(res)(f32)
//   K2 conv+silu                   -> u(f32) + u_bf
//   K3 gemm<3>: x_dbl = u @ W_x    -> xdbl(f32, scan) + xdbl_bf (N=96, guarded)
//   K4 gemm<2>: delta = softplus(xdbl[:, :64] @ W_dt + b)
//   K5 scan: h = exp(dt*A)h + dt*B*u ; y = <h,C> + u*D ; y *= silu(res) -> y2_bf
//   K6 gemm<0>: out = y2 @ W_out   -> d_out (f32)
// GEMM: 128x128 block, 8 waves x (32x64), double-buffered LDS, bf16 operands:
// A-tile fill is a pure b128 copy (no VALU); B-tile transposes k-pairs with
// 4 shift/mask ops per dword. Fragments follow the ISA 16-bit A/B striping
// (two contiguous 16B LDS loads each).
// ---------------------------------------------------------------------------

typedef __attribute__((ext_vector_type(16))) __bf16 v16bf;
typedef __attribute__((ext_vector_type(8)))  float  v8f;

union Frag16 { v16bf v; uint4 q[2]; };

__device__ __forceinline__ unsigned short f2bf(float f) {
  __bf16 h = (__bf16)f;                       // native cvt on gfx1250
  return __builtin_bit_cast(unsigned short, h);
}
__device__ __forceinline__ unsigned pack2bf(float lo, float hi) {
  return (unsigned)f2bf(lo) | ((unsigned)f2bf(hi) << 16);
}
__device__ __forceinline__ float sigmoidf_(float x) { return 1.0f / (1.0f + __expf(-x)); }

#define BM 128
#define BN 128
#define BK 32

// f32 -> bf16 streaming convert, 4 elements/thread.
__global__ __launch_bounds__(256)
void cvt_f32_bf16(const float* __restrict__ src, unsigned short* __restrict__ dst, long n4) {
  long i = (long)blockIdx.x * 256 + threadIdx.x;
  if (i >= n4) return;
  float4 v = ((const float4*)src)[i];
  uint2 w;
  w.x = pack2bf(v.x, v.y);
  w.y = pack2bf(v.z, v.w);
  ((uint2*)dst)[i] = w;
}

// C[M,N] = A[M,K] @ B[K,N]; bf16 in, f32 WMMA accumulate.
// Requires M % BM == 0, K % BK == 0, lda/ldb multiples of 8 (true for all calls).
// GUARD bounds-checks N only (needed just for the N=96 GEMM).
// EPI 0: C = v     EPI 1: split -> C | silu -> C2 (each width N/2)
// EPI 2: C = softplus(v + bias[col])     EPI 3: C = v and Cbf = bf16(v)
template<int EPI, bool GUARD>
__global__ __launch_bounds__(256)
void gemm_wmma_bf16(const unsigned short* __restrict__ A, const unsigned short* __restrict__ B,
                    float* __restrict__ C, float* __restrict__ C2,
                    unsigned short* __restrict__ Cbf, const float* __restrict__ bias,
                    int M, int N, int K, int lda, int ldb, int ldc)
{
  __shared__ __align__(16) unsigned short As[2][BM * BK];   // [m][k]
  __shared__ __align__(16) unsigned short Bs[2][BN * BK];   // [n][k] (K-contiguous)

  const int tid  = threadIdx.x;
  const int wave = tid >> 5;
  const int lane = tid & 31;
  const int half = lane >> 4;       // ISA lane-group g
  const int ln   = lane & 15;

  const int wm = wave >> 1;         // 0..3  (32-row wave tile)
  const int wn = wave & 1;          // 0..1  (64-col wave tile)
  const int blockM = blockIdx.x * BM;
  const int blockN = blockIdx.y * BN;

  v8f acc[2][4] = {{}, {}};

  // A: 2 x 16B chunks/thread (8 bf16 each); 4 chunks per 32-wide row.
  // B: one (k, k+1) row pair x 8 columns per thread (two b128 loads).
  uint4 aReg[2], b0Reg, b1Reg;

  auto loadTile = [&](int k0) {
    #pragma unroll
    for (int i = 0; i < 2; ++i) {
      int e = tid + i * 256;
      int row = e >> 2, c16 = e & 3;
      aReg[i] = *(const uint4*)&A[(blockM + row) * lda + k0 + c16 * 8];
    }
    int kp = tid >> 4, ng = tid & 15;
    int k = kp * 2, gn = blockN + ng * 8;
    if (!GUARD || (gn + 7 < N)) {
      b0Reg = *(const uint4*)&B[(k0 + k    ) * ldb + gn];
      b1Reg = *(const uint4*)&B[(k0 + k + 1) * ldb + gn];
    } else {
      b0Reg = make_uint4(0u, 0u, 0u, 0u);
      b1Reg = make_uint4(0u, 0u, 0u, 0u);
    }
  };

  auto storeTile = [&](int bi) {
    #pragma unroll
    for (int i = 0; i < 2; ++i) {
      int e = tid + i * 256;
      int row = e >> 2, c16 = e & 3;
      *(uint4*)&As[bi][row * BK + c16 * 8] = aReg[i];      // pure copy, no VALU
    }
    int kp = tid >> 4, ng = tid & 15;
    int k = kp * 2;
    unsigned r0[4] = {b0Reg.x, b0Reg.y, b0Reg.z, b0Reg.w};
    unsigned r1[4] = {b1Reg.x, b1Reg.y, b1Reg.z, b1Reg.w};
    #pragma unroll
    for (int c = 0; c < 4; ++c) {                           // transpose k-pairs
      unsigned lo = (r0[c] & 0xffffu) | (r1[c] << 16);      // {k,n} {k+1,n}
      unsigned hi = (r0[c] >> 16) | (r1[c] & 0xffff0000u);  // {k,n+1} {k+1,n+1}
      *(unsigned*)&Bs[bi][(ng * 8 + c * 2    ) * BK + k] = lo;
      *(unsigned*)&Bs[bi][(ng * 8 + c * 2 + 1) * BK + k] = hi;
    }
  };

  auto computeTile = [&](int bi) {
    Frag16 a[2], b[4];
    #pragma unroll
    for (int i = 0; i < 2; ++i) {
      int m = wm * 32 + i * 16 + ln;                              // lanes 0-15/16-31: same M
      a[i].q[0] = *(const uint4*)&As[bi][m * BK + 8 * half];      // K = 8g..8g+7
      a[i].q[1] = *(const uint4*)&As[bi][m * BK + 16 + 8 * half]; // K = 16+8g..23+8g
    }
    #pragma unroll
    for (int j = 0; j < 4; ++j) {
      int n = wn * 64 + j * 16 + ln;
      const uint4* p = (const uint4*)&Bs[bi][n * BK + 8 * half]; // K = 8g..8g+7, 16+8g..23+8g
      b[j].q[0] = p[0];
      b[j].q[1] = p[2];
    }
    #pragma unroll
    for (int i = 0; i < 2; ++i)
      #pragma unroll
      for (int j = 0; j < 4; ++j)
        acc[i][j] = __builtin_amdgcn_wmma_f32_16x16x32_bf16(
            false, a[i].v, false, b[j].v, (short)0, acc[i][j], false, false);
  };

  // ---- software-pipelined K loop: load(t+1) || compute(t), 1 barrier/step ----
  int buf = 0;
  loadTile(0);
  storeTile(0);
  __syncthreads();
  for (int k0 = 0; k0 < K; k0 += BK) {
    const bool hasNext = (k0 + BK < K);
    if (hasNext) {
      loadTile(k0 + BK);                       // global b128 loads in flight
      if (k0 + 2 * BK < K) {                   // cache prefetch one tile further
        __builtin_prefetch(&A[(blockM + (tid >> 1)) * lda + k0 + 2 * BK], 0, 1);
        __builtin_prefetch(&B[(k0 + 2 * BK + (tid >> 7)) * ldb + blockN + (tid & 127)], 0, 1);
      }
    }
    computeTile(buf);                          // ds_load_b128 -> v_wmma chain
    if (hasNext) storeTile(buf ^ 1);
    __syncthreads();
    buf ^= 1;
  }

  // ---- epilogue: C/D layout -> row = r + 8*half, col = ln ----
  #pragma unroll
  for (int i = 0; i < 2; ++i)
    #pragma unroll
    for (int j = 0; j < 4; ++j)
      #pragma unroll
      for (int r = 0; r < 8; ++r) {
        int gr = blockM + wm * 32 + i * 16 + r + 8 * half;
        int gc = blockN + wn * 64 + j * 16 + ln;
        if (GUARD && gc >= N) continue;
        float v = acc[i][j][r];
        if (EPI == 0 || EPI == 3) {
          C[gr * ldc + gc] = v;
          if (EPI == 3) Cbf[gr * ldc + gc] = f2bf(v);
        } else if (EPI == 1) {
          int hw = N >> 1;
          if (gc < hw) C[gr * hw + gc] = v;                        // xs (pre-conv)
          else         C2[gr * hw + (gc - hw)] = v * sigmoidf_(v); // silu(res)
        } else { // EPI == 2
          v += bias[gc];
          C[gr * ldc + gc] = (v > 20.0f) ? v : __logf(1.0f + __expf(v));
        }
      }
}

// depthwise causal conv1d (kernel 4, left pad 3) + bias + SiLU -> u (f32 + bf16)
__global__ __launch_bounds__(256)
void conv_silu_kernel(const float* __restrict__ xs, const float* __restrict__ w,
                      const float* __restrict__ bias, float* __restrict__ u,
                      unsigned short* __restrict__ ubf,
                      int Bb, int Ll, int Din)
{
  long idx = (long)blockIdx.x * 256 + threadIdx.x;
  long total = (long)Bb * Ll * Din;
  if (idx >= total) return;
  int d = (int)(idx % Din);
  long bl = idx / Din;
  int l = (int)(bl % Ll);
  int b = (int)(bl / Ll);
  float acc = bias[d];
  #pragma unroll
  for (int k = 0; k < 4; ++k) {
    int li = l + k - 3;
    if (li >= 0) acc += w[d * 4 + k] * xs[((long)b * Ll + li) * Din + d];
  }
  float s = acc * sigmoidf_(acc);
  u[idx] = s;
  ubf[idx] = f2bf(s);
}

// Selective scan: each half-wave owns channel d; lane n = lane%16 owns state (d,n).
// h lives in a register; <h,C> reduced over the 16 n-lanes via shuffles per step.
// hs (512 MB if materialized) is never written to memory. Output y2 in bf16
// (it is only consumed as the A operand of the final WMMA GEMM).
__global__ __launch_bounds__(256)
void scan_kernel(const float* __restrict__ delta, const float* __restrict__ u,
                 const float* __restrict__ xdbl, const float* __restrict__ A_log,
                 const float* __restrict__ Dp, const float* __restrict__ sres,
                 unsigned short* __restrict__ y2bf,
                 int Bb, int Ll, int Din, int Nn, int ldx, int dtr)
{
  const int tid  = threadIdx.x;
  const int wave = tid >> 5;
  const int lane = tid & 31;
  const int half = lane >> 4;
  const int n    = lane & 15;
  const int pair = blockIdx.x * 8 + wave;     // one wave handles 2 channels d
  const int d = (pair * 2 + half) % Din;
  const int b = (pair * 2) / Din;

  const float Aval = -__expf(A_log[d * Nn + n]);
  const float Dval = Dp[d];
  float h = 0.0f;

  for (int l = 0; l < Ll; ++l) {
    const long t  = (long)b * Ll + l;
    const float dv = delta[t * Din + d];
    const float uv = u[t * Din + d];
    const float Bv = xdbl[t * ldx + dtr + n];
    const float Cv = xdbl[t * ldx + dtr + Nn + n];
    const float dA = __expf(dv * Aval);
    h = dA * h + dv * Bv * uv;
    float c = h * Cv;
    c += __shfl_xor(c, 1, 16);
    c += __shfl_xor(c, 2, 16);
    c += __shfl_xor(c, 4, 16);
    c += __shfl_xor(c, 8, 16);
    if (n == 0) {
      float y = c + uv * Dval;
      y2bf[t * Din + d] = f2bf(y * sres[t * Din + d]);
    }
  }
}

extern "C" void kernel_launch(void* const* d_in, const int* in_sizes, int n_in,
                              void* d_out, int out_size, void* d_ws, size_t ws_size,
                              hipStream_t stream) {
  const float* x     = (const float*)d_in[0];
  const float* W_in  = (const float*)d_in[1];
  const float* convw = (const float*)d_in[2];
  const float* convb = (const float*)d_in[3];
  const float* W_x   = (const float*)d_in[4];
  const float* W_dt  = (const float*)d_in[5];
  const float* b_dt  = (const float*)d_in[6];
  const float* W_out = (const float*)d_in[7];
  const float* A_log = (const float*)d_in[8];
  const float* Dp    = (const float*)d_in[9];

  const int Bb = 2, Ll = 2048, DIM = 1024, DIN = 2048, Nn = 16, DTR = 64;
  const int XDBL = DTR + 2 * Nn;        // 96
  const int ML = Bb * Ll;               // 4096

  // ---- workspace layout: f32 region first, then 16B-aligned bf16 region ----
  float* wsf   = (float*)d_ws;
  float* xs    = wsf; wsf += (size_t)ML * DIN;
  float* sres  = wsf; wsf += (size_t)ML * DIN;
  float* u     = wsf; wsf += (size_t)ML * DIN;
  float* delta = wsf; wsf += (size_t)ML * DIN;
  float* xdbl  = wsf; wsf += (size_t)ML * XDBL;

  unsigned short* wsh  = (unsigned short*)wsf;
  unsigned short* x_bf    = wsh; wsh += (size_t)ML * DIM;
  unsigned short* Win_bf  = wsh; wsh += (size_t)DIM * 2 * DIN;
  unsigned short* Wx_bf   = wsh; wsh += (size_t)DIN * XDBL;
  unsigned short* Wdt_bf  = wsh; wsh += (size_t)DTR * DIN;
  unsigned short* Wout_bf = wsh; wsh += (size_t)DIN * DIM;
  unsigned short* u_bf    = wsh; wsh += (size_t)ML * DIN;
  unsigned short* xdbl_bf = wsh; wsh += (size_t)ML * XDBL;
  unsigned short* y2_bf   = wsh; wsh += (size_t)ML * DIN;

  dim3 blk(256);
  auto cvt = [&](const float* s, unsigned short* dst, long n) {
    long n4 = n / 4;
    cvt_f32_bf16<<<(unsigned)((n4 + 255) / 256), blk, 0, stream>>>(s, dst, n4);
  };

  // C0: one-shot precision conversion of inputs/weights
  cvt(x,     x_bf,    (long)ML * DIM);
  cvt(W_in,  Win_bf,  (long)DIM * 2 * DIN);
  cvt(W_x,   Wx_bf,   (long)DIN * XDBL);
  cvt(W_dt,  Wdt_bf,  (long)DTR * DIN);
  cvt(W_out, Wout_bf, (long)DIN * DIM);

  // K1: xr = x @ W_in -> xs | silu(res)   (all tile-aligned)
  dim3 g1(ML / BM, (2 * DIN) / BN);
  gemm_wmma_bf16<1, false><<<g1, blk, 0, stream>>>(x_bf, Win_bf, xs, sres, nullptr, nullptr,
      ML, 2 * DIN, DIM, DIM, 2 * DIN, 0);

  // K2: depthwise conv + SiLU -> u (f32 + bf16)
  long total = (long)ML * DIN;
  conv_silu_kernel<<<(unsigned)((total + 255) / 256), blk, 0, stream>>>(
      xs, convw, convb, u, u_bf, Bb, Ll, DIN);

  // K3: x_dbl = u @ W_x -> f32 (scan) + bf16 (next GEMM's A)   (N=96 -> GUARD)
  dim3 g3(ML / BM, 1);
  gemm_wmma_bf16<3, true><<<g3, blk, 0, stream>>>(u_bf, Wx_bf, xdbl, nullptr, xdbl_bf, nullptr,
      ML, XDBL, DIN, DIN, XDBL, XDBL);

  // K4: delta = softplus(x_dbl[:, :64] @ W_dt + b_dt)   (K=64 = 2*BK, aligned)
  dim3 g4(ML / BM, DIN / BN);
  gemm_wmma_bf16<2, false><<<g4, blk, 0, stream>>>(xdbl_bf, Wdt_bf, delta, nullptr, nullptr, b_dt,
      ML, DIN, DTR, XDBL, DIN, DIN);

  // K5: selective scan -> y2 (bf16)
  int nwaves = Bb * DIN / 2;            // 2048 waves, 8 per block
  scan_kernel<<<nwaves / 8, blk, 0, stream>>>(delta, u, xdbl, A_log, Dp, sres, y2_bf,
      Bb, Ll, DIN, Nn, XDBL, DTR);

  // K6: out = y2 @ W_out
  dim3 g6(ML / BM, DIM / BN);
  gemm_wmma_bf16<0, false><<<g6, blk, 0, stream>>>(y2_bf, Wout_bf, (float*)d_out, nullptr, nullptr, nullptr,
      ML, DIM, DIN, DIN, DIM, DIM);
}
